// SatCNFEvaluator_31353261260818
// MI455X (gfx1250) — compile-verified, matrix operand order checked by reading, not run on hardware
//
#include <hip/hip_runtime.h>

// ---------------------------------------------------------------------------
// SAT CNF evaluator for MI455X (gfx1250, wave32).
//
// Pipeline (all on `stream`, graph-capture safe):
//   memset  : zero d_out (clause region + heads) and d_ws accumulators
//   kernel 1: edge phase  — gather pred, threshold, idempotent scatter of 1.0
//   kernel 2: clause phase — read clause_values, 4-bin counts reduced with
//             v_wmma_f32_16x16x32_f16 cross-lane reductions + LDS/global atomics
//   kernel 3: finalize    — sat_flag and (max_sat - batch_values) for B=4
// ---------------------------------------------------------------------------

typedef __attribute__((ext_vector_type(16))) _Float16 v16h;
typedef __attribute__((ext_vector_type(8)))  float    v8f;

// Exact 32-lane sum of an integer-valued float (|x| <= 256) using the matrix
// unit. Result is broadcast to every lane.
//
// WMMA#1: A[m,k] has lane m's value in one K slot (lanes 0-15 -> rows 0-15,
// lanes 16-31 -> rows 0-15 again per the 16-bit A striping), B = ones, so
// D[m,*] = x[m] + x[m+16] = S[m], with D VGPR r holding S[r] (lanes 0-15) and
// S[r+8] (lanes 16-31).  Adding DV0..3 and DV4..7 gives two per-lane partials
// whose lane/half structure covers {S[0..3],S[8..11]} and {S[4..7],S[12..15]}.
// WMMA#2: A = ones so D2[m,n] = sum_k B[k,n]; column n of B draws all 16
// halves of lanes n and n+16, i.e. exactly the four partials -> total sum.
// All intermediates are integers <= 2048, exact in f16; accumulation is f32.
__device__ __forceinline__ float wave_sum32(float x, v16h ones) {
  v16h a;
#pragma unroll
  for (int k = 0; k < 16; ++k) a[k] = (_Float16)0.0f;
  a[0] = (_Float16)x;
  v8f c;
#pragma unroll
  for (int k = 0; k < 8; ++k) c[k] = 0.0f;
  v8f d = __builtin_amdgcn_wmma_f32_16x16x32_f16(false, a, false, ones,
                                                 (short)0, c, false, false);
  float t0 = (d[0] + d[1]) + (d[2] + d[3]);
  float t1 = (d[4] + d[5]) + (d[6] + d[7]);
  v16h b;
#pragma unroll
  for (int k = 0; k < 16; ++k) b[k] = (_Float16)0.0f;
  b[0] = (_Float16)t0;
  b[1] = (_Float16)t1;
  v8f d2 = __builtin_amdgcn_wmma_f32_16x16x32_f16(false, ones, false, b,
                                                  (short)0, c, false, false);
  return d2[0];
}

// ---------------------------------------------------------------------------
// Kernel 1: edges. Streams graph_map + edge_feature with B128 loads, gathers
// pred (2 MB, L2-resident), and does an idempotent scatter: a clause is
// satisfied iff ANY literal is satisfied, so racing stores of 1.0 are benign.
// Math follows the reference exactly: ev = e*p + (1-e)*0.5 ; sat = ev > 0.5.
// ---------------------------------------------------------------------------
__global__ void sat_edge_kernel(const float* __restrict__ pred,
                                const float* __restrict__ ef,
                                const int*   __restrict__ vidx,
                                const int*   __restrict__ fidx,
                                float*       __restrict__ clause_out,
                                int n4, int nE) {
  int i = blockIdx.x * blockDim.x + threadIdx.x;
  if (i < n4) {
    int4   v = ((const int4*)vidx)[i];
    int4   f = ((const int4*)fidx)[i];
    float4 e = ((const float4*)ef)[i];
    float p0 = pred[v.x], p1 = pred[v.y], p2 = pred[v.z], p3 = pred[v.w];
    float s0 = e.x * p0 + (1.0f - e.x) * 0.5f;
    float s1 = e.y * p1 + (1.0f - e.y) * 0.5f;
    float s2 = e.z * p2 + (1.0f - e.z) * 0.5f;
    float s3 = e.w * p3 + (1.0f - e.w) * 0.5f;
    if (s0 > 0.5f) clause_out[f.x] = 1.0f;
    if (s1 > 0.5f) clause_out[f.y] = 1.0f;
    if (s2 > 0.5f) clause_out[f.z] = 1.0f;
    if (s3 > 0.5f) clause_out[f.w] = 1.0f;
  }
  // scalar tail for E % 4 (zero for E = 6M, kept for generality)
  int tail = n4 << 2;
  int rem  = nE - tail;
  if (blockIdx.x == 0 && (int)threadIdx.x < rem) {
    int j = tail + threadIdx.x;
    float e = ef[j];
    float p = pred[vidx[j]];
    float s = e * p + (1.0f - e) * 0.5f;
    if (s > 0.5f) clause_out[fidx[j]] = 1.0f;
  }
}

// ---------------------------------------------------------------------------
// Kernel 2: clauses. clause_out already holds clause_values (0.0/1.0) in the
// output buffer. Accumulate per-thread 4-bin counts (satisfied clauses and
// total clauses per batch), reduce each across the wave with the WMMA
// reduction, then LDS-atomic per block and one global atomic per bin per
// block into cacheline-padded accumulators.
// ITEMS=16 per thread -> per-lane counters <= 16 (exact in f16).
// ---------------------------------------------------------------------------
#define CL_ITEMS 16

__global__ void sat_clause_kernel(const float* __restrict__ cv,
                                  const int*   __restrict__ bfm,
                                  float*       __restrict__ accum,
                                  int nF) {
  __shared__ float sacc[8];
  if (threadIdx.x < 8) sacc[threadIdx.x] = 0.0f;
  __syncthreads();

  float c0 = 0.f, c1 = 0.f, c2 = 0.f, c3 = 0.f;  // satisfied per batch
  float t0 = 0.f, t1 = 0.f, t2 = 0.f, t3 = 0.f;  // clause count per batch
  int base = blockIdx.x * (blockDim.x * CL_ITEMS);
#pragma unroll
  for (int it = 0; it < CL_ITEMS; ++it) {
    int i = base + it * blockDim.x + threadIdx.x;
    if (i < nF) {
      float c = cv[i];
      int   b = bfm[i];
      c0 += (b == 0) ? c : 0.0f;  t0 += (b == 0) ? 1.0f : 0.0f;
      c1 += (b == 1) ? c : 0.0f;  t1 += (b == 1) ? 1.0f : 0.0f;
      c2 += (b == 2) ? c : 0.0f;  t2 += (b == 2) ? 1.0f : 0.0f;
      c3 += (b == 3) ? c : 0.0f;  t3 += (b == 3) ? 1.0f : 0.0f;
    }
  }

  // Uniform control flow from here down: EXEC is all-ones for the WMMAs.
  v16h ones;
#pragma unroll
  for (int k = 0; k < 16; ++k) ones[k] = (_Float16)1.0f;

  float r0 = wave_sum32(c0, ones);
  float r1 = wave_sum32(c1, ones);
  float r2 = wave_sum32(c2, ones);
  float r3 = wave_sum32(c3, ones);
  float r4 = wave_sum32(t0, ones);
  float r5 = wave_sum32(t1, ones);
  float r6 = wave_sum32(t2, ones);
  float r7 = wave_sum32(t3, ones);

  if ((threadIdx.x & 31) == 0) {
    atomicAdd(&sacc[0], r0);
    atomicAdd(&sacc[1], r1);
    atomicAdd(&sacc[2], r2);
    atomicAdd(&sacc[3], r3);
    atomicAdd(&sacc[4], r4);
    atomicAdd(&sacc[5], r5);
    atomicAdd(&sacc[6], r6);
    atomicAdd(&sacc[7], r7);
  }
  __syncthreads();
  if (threadIdx.x < 8) {
    // accumulators padded to 128B (32 floats) to avoid cacheline contention
    atomicAdd(&accum[threadIdx.x * 32], sacc[threadIdx.x]);
  }
}

// ---------------------------------------------------------------------------
// Kernel 3: finalize. accum[b*32] = batch_values[b], accum[(4+b)*32] =
// max_sat[b]. Both are exact integers in f32, so == compare matches the
// reference's (max_sat == batch_values).
// ---------------------------------------------------------------------------
__global__ void sat_finalize_kernel(const float* __restrict__ accum,
                                    float* __restrict__ out, int Bn) {
  int b = threadIdx.x;
  if (b < Bn) {
    float bv = accum[b * 32];
    float ms = accum[(b + 4) * 32];
    out[b]      = (ms == bv) ? 1.0f : 0.0f;  // sat_flag
    out[Bn + b] = ms - bv;                   // unsatisfied clause count
  }
}

extern "C" void kernel_launch(void* const* d_in, const int* in_sizes, int n_in,
                              void* d_out, int out_size, void* d_ws, size_t ws_size,
                              hipStream_t stream) {
  const float* pred = (const float*)d_in[0];  // [V,1] f32
  const float* ef   = (const float*)d_in[1];  // [E,1] f32 (+1/-1)
  const int*   gmap = (const int*)d_in[2];    // [2,E] int32 (row0=var, row1=fun)
  // d_in[3] (batch_variable_map) is unused by the reference computation.
  const int*   bfm  = (const int*)d_in[4];    // [F] int32, sorted

  const int nE = in_sizes[1];
  const int nF = in_sizes[4];
  const int Bn = (out_size - nF) / 2;         // = 4

  float* out        = (float*)d_out;
  float* clause_out = out + 2 * Bn;           // clause_values live in d_out
  float* accum      = (float*)d_ws;           // 8 padded f32 accumulators

  // Zero output (clause region must start at 0; heads overwritten later) and
  // accumulators. hipMemsetAsync is legal under graph capture.
  hipMemsetAsync(d_out, 0, (size_t)out_size * sizeof(float), stream);
  hipMemsetAsync(d_ws, 0, 256 * sizeof(float), stream);

  const int* vidx = gmap;
  const int* fidx = gmap + nE;

  const int threads = 256;
  int n4 = nE >> 2;
  int gE = (n4 + threads - 1) / threads;
  if (gE < 1) gE = 1;
  sat_edge_kernel<<<gE, threads, 0, stream>>>(pred, ef, vidx, fidx,
                                              clause_out, n4, nE);

  int perBlock = threads * CL_ITEMS;
  int gF = (nF + perBlock - 1) / perBlock;
  sat_clause_kernel<<<gF, threads, 0, stream>>>(clause_out, bfm, accum, nF);

  sat_finalize_kernel<<<1, 32, 0, stream>>>(accum, out, Bn);
}
